// LabelGCNAttentionRNNv5_56246891709054
// MI455X (gfx1250) — matrix operand order repaired and checked by reading, not compiled, not run on hardware
//
#include <hip/hip_runtime.h>
#include <hip/hip_bf16.h>

// ---------------------------------------------------------------------------
// GIN(2 layers, D=128) + residual MLPs + 1024-label head for gfx1250 (MI455X).
// Matrix math: v_wmma_f32_16x16x32_bf16 (f32 accumulate) with bf16 activations
// staged in memory; GEMM inner loop is software-pipelined loads + WMMA with a
// parity double buffer so every WMMA overlaps the next fragment's loads.
// Graph aggregation: L2-resident global fp32 atomics (dest fits in 192MB L2).
// ---------------------------------------------------------------------------

typedef __attribute__((ext_vector_type(16))) __bf16 v16bf;
typedef __attribute__((ext_vector_type(2)))  __bf16 v2bf;
typedef __attribute__((ext_vector_type(8)))  float  v8f;

#define DHID 128

static __device__ __forceinline__ unsigned short f2bf_rne(float x) {
  union { float f; unsigned u; } a; a.f = x;
  unsigned r = a.u + 0x7FFFu + ((a.u >> 16) & 1u);   // round to nearest even
  return (unsigned short)(r >> 16);
}

static __device__ __forceinline__ unsigned int pk_bf16(float lo, float hi) {
#if __has_builtin(__builtin_amdgcn_cvt_pk_bf16_f32)
  union { v2bf v; unsigned int u; } c;
  c.v = __builtin_amdgcn_cvt_pk_bf16_f32(lo, hi);
  return c.u;
#else
  return (unsigned)f2bf_rne(lo) | ((unsigned)f2bf_rne(hi) << 16);
#endif
}

union BF16Frag {
  v16bf v;
  unsigned short u[16];
  unsigned int w[8];
};

static __device__ __forceinline__ BF16Frag load_frag(const uint4* __restrict__ p) {
  uint4 q0 = p[0];
  uint4 q1 = p[1];
  BF16Frag f;
  f.w[0] = q0.x; f.w[1] = q0.y; f.w[2] = q0.z; f.w[3] = q0.w;
  f.w[4] = q1.x; f.w[5] = q1.y; f.w[6] = q1.z; f.w[7] = q1.w;
  return f;
}

// A fragment per ISA 16-bit A 16x32 layout: lane half h, row = lane&15,
// holds K = k0..k0+7 and K = k0+16..k0+23 where k0 = kk*32 + h*8.
static __device__ __forceinline__ BF16Frag load_a(const unsigned short* __restrict__ ap,
                                                  int k0) {
  uint4 qa = *(const uint4*)(ap + k0);
  uint4 qb = *(const uint4*)(ap + k0 + 16);
  BF16Frag f;
  f.w[0] = qa.x; f.w[1] = qa.y; f.w[2] = qa.z; f.w[3] = qa.w;
  f.w[4] = qb.x; f.w[5] = qb.y; f.w[6] = qb.z; f.w[7] = qb.w;
  return f;
}

// ---------------------------------------------------------------------------
// Repack W [128 x N] fp32 row-major into per-wave bf16 B fragments.
// Fragment f = kt*ntiles + nt ; lane l holds col n = nt*16 + (l&15),
// K = kt*32 + (l>>4)*8 + (j&7) + (j>>3)*16  (j = 0..15), 16 bf16 per lane.
// ---------------------------------------------------------------------------
__global__ void pack_b_kernel(const float* __restrict__ W,
                              unsigned short* __restrict__ F, int N) {
  int idx = blockIdx.x * blockDim.x + threadIdx.x;
  int total = DHID * N;                       // 4 kt * (N/16) nt * 32 * 16
  if (idx >= total) return;
  int j      = idx & 15;
  int lane   = (idx >> 4) & 31;
  int f      = idx >> 9;
  int ntiles = N >> 4;
  int nt     = f % ntiles;
  int kt     = f / ntiles;
  int n      = nt * 16 + (lane & 15);
  int k      = kt * 32 + (lane >> 4) * 8 + (j & 7) + (j >> 3) * 16;
  F[idx] = f2bf_rne(W[(size_t)k * N + n]);
}

// out = (1+eps)*in   (float4-vectorized; n4 = element-count/4)
__global__ void scale_copy_kernel(const float4* __restrict__ in,
                                  const float* __restrict__ eps,
                                  float4* __restrict__ out, int n4) {
  int i = blockIdx.x * blockDim.x + threadIdx.x;
  if (i >= n4) return;
  float a = 1.0f + eps[0];
  float4 v = in[i];
  v.x *= a; v.y *= a; v.z *= a; v.w *= a;
  out[i] = v;
}

// bf16 row-major mirror of an fp32 tensor (hardware packed cvt when available)
__global__ void to_bf16_kernel(const float4* __restrict__ in,
                               uint2* __restrict__ out, int n4) {
  int i = blockIdx.x * blockDim.x + threadIdx.x;
  if (i >= n4) return;
  float4 v = in[i];
  uint2 o;
  o.x = pk_bf16(v.x, v.y);
  o.y = pk_bf16(v.z, v.w);
  out[i] = o;
}

// X[dst] += H[src] : one wave per edge, 4 floats per lane (L2-resident atomics)
__global__ void scatter_add_kernel(const float* __restrict__ H,
                                   const long long* __restrict__ src,
                                   const long long* __restrict__ dst,
                                   float* __restrict__ X, int nEdges) {
  int gid = blockIdx.x * blockDim.x + threadIdx.x;
  int e = gid >> 5;
  if (e >= nEdges) return;
  int lane = gid & 31;
  long long s = src[e];
  long long d = dst[e];
  const float4 v = *(const float4*)(H + (size_t)s * DHID + lane * 4);
  float* p = X + (size_t)d * DHID + lane * 4;
  __hip_atomic_fetch_add(p + 0, v.x, __ATOMIC_RELAXED, __HIP_MEMORY_SCOPE_AGENT);
  __hip_atomic_fetch_add(p + 1, v.y, __ATOMIC_RELAXED, __HIP_MEMORY_SCOPE_AGENT);
  __hip_atomic_fetch_add(p + 2, v.z, __ATOMIC_RELAXED, __HIP_MEMORY_SCOPE_AGENT);
  __hip_atomic_fetch_add(p + 3, v.w, __ATOMIC_RELAXED, __HIP_MEMORY_SCOPE_AGENT);
}

// ---------------------------------------------------------------------------
// WMMA GEMM: epilogue(Abf[M x 128] @ W + bias)
//   one wave owns a 16 x (16*T) output strip; A fragment reused across T
//   WMMAs. A is prefetched one k-step ahead; B uses a parity-indexed double
//   buffer (distinct register blocks, overlapping live ranges) so each WMMA
//   issues while the next B fragment's loads are in flight.
// MODE 0: v=relu(t)            -> C fp32 + Cbf bf16   (feeds residual GEMM)
// MODE 1: v=R + relu(t)        -> C fp32              (layer output H1)
// MODE 2: v=.5*(P+R+relu(t))   -> Cbf bf16 only       (pooled, feeds head)
// MODE 3: v=t                  -> C fp32              (label head)
// ---------------------------------------------------------------------------
template <int MODE, int T>
__global__ void gemm_wmma_kernel(const unsigned short* __restrict__ Abf,
                                 const unsigned short* __restrict__ Bfrag,
                                 const float* __restrict__ bias,
                                 const float* __restrict__ R,
                                 const float* __restrict__ P,
                                 float* __restrict__ C,
                                 unsigned short* __restrict__ Cbf,
                                 int M, int N) {
  const int lane    = threadIdx.x & 31;
  const int wave    = threadIdx.x >> 5;
  const int ntiles  = N >> 4;
  const int mtiles  = (M + 15) >> 4;
  const int ngroups = ntiles / T;
  const int task = blockIdx.x * 8 + wave;           // 8 waves per block
  if (task >= mtiles * ngroups) return;             // wave-uniform exit
  const int mt  = task / ngroups;
  const int nt0 = (task % ngroups) * T;
  const int mbase = mt * 16;
  const int half  = lane >> 4;

  int arow = mbase + (lane & 15);
  if (arow >= M) arow = M - 1;                      // clamp, stores are guarded
  const unsigned short* __restrict__ ap = Abf + (size_t)arow * DHID;
  const uint4* __restrict__ bb = (const uint4*)Bfrag + (size_t)lane * 2;

  v8f acc[T];
#pragma unroll
  for (int t = 0; t < T; ++t) acc[t] = {};

  // ---- software-pipelined K loop (K = 128 = 4 x 32) ----
  BF16Frag a_cur = load_a(ap, half * 8);
  BF16Frag bbuf[2];
  bbuf[0] = load_frag(bb + (size_t)nt0 * 64);
#pragma unroll
  for (int kk = 0; kk < 4; ++kk) {
    BF16Frag a_next = a_cur;
    if (kk < 3) a_next = load_a(ap, (kk + 1) * 32 + half * 8);
#pragma unroll
    for (int t = 0; t < T; ++t) {
      const int cur = (kk * T + t) & 1;             // constant after unroll
      if (t + 1 < T)
        bbuf[cur ^ 1] =
            load_frag(bb + ((size_t)kk * ntiles + nt0 + t + 1) * 64);
      else if (kk < 3)
        bbuf[cur ^ 1] =
            load_frag(bb + ((size_t)(kk + 1) * ntiles + nt0) * 64);
      acc[t] = __builtin_amdgcn_wmma_f32_16x16x32_bf16(
          false, a_cur.v, false, bbuf[cur].v, (short)0, acc[t], false, false);
    }
    a_cur = a_next;
  }

  // ---- epilogue: C/D layout lane(0-15):M=r, lane(16-31):M=8+r, N=lane&15 ----
  const int rbase = mbase + half * 8;
#pragma unroll
  for (int t = 0; t < T; ++t) {
    const int col = (nt0 + t) * 16 + (lane & 15);
    const float bcol = bias[col];
#pragma unroll
    for (int r = 0; r < 8; ++r) {
      const int row = rbase + r;
      if (row < M) {
        float tv = acc[t][r] + bcol;
        if (MODE == 0) {
          float v = fmaxf(tv, 0.0f);
          C[(size_t)row * N + col]   = v;
          Cbf[(size_t)row * N + col] = f2bf_rne(v);
        } else if (MODE == 1) {
          C[(size_t)row * N + col] =
              R[(size_t)row * DHID + col] + fmaxf(tv, 0.0f);
        } else if (MODE == 2) {
          float v = 0.5f * (P[(size_t)row * DHID + col] +
                            R[(size_t)row * DHID + col] + fmaxf(tv, 0.0f));
          Cbf[(size_t)row * DHID + col] = f2bf_rne(v);
        } else {
          C[(size_t)row * N + col] = tv;
        }
      }
    }
  }
}

// ---------------------------------------------------------------------------
extern "C" void kernel_launch(void* const* d_in, const int* in_sizes, int n_in,
                              void* d_out, int out_size, void* d_ws, size_t ws_size,
                              hipStream_t stream) {
  const float*     inputs = (const float*)d_in[0];
  const long long* src    = (const long long*)d_in[1];
  const long long* dst    = (const long long*)d_in[2];
  const float*     eps1   = (const float*)d_in[3];
  const float*     W1     = (const float*)d_in[4];
  const float*     b1     = (const float*)d_in[5];
  const float*     rW1    = (const float*)d_in[6];
  const float*     rb1    = (const float*)d_in[7];
  const float*     eps2   = (const float*)d_in[8];
  const float*     W2     = (const float*)d_in[9];
  const float*     b2     = (const float*)d_in[10];
  const float*     rW2    = (const float*)d_in[11];
  const float*     rb2    = (const float*)d_in[12];
  const float*     linW   = (const float*)d_in[13];
  const float*     linb   = (const float*)d_in[14];

  const int M  = in_sizes[0] / DHID;   // 100000 nodes
  const int E  = in_sizes[1];          // 1.6M edges
  const int NL = in_sizes[14];         // 1024 labels

  // ---- workspace layout ----
  char* ws = (char*)d_ws;
  const size_t szX  = (size_t)M * DHID * sizeof(float);
  const size_t szXb = (size_t)M * DHID * sizeof(unsigned short);
  float*          X     = (float*)(ws);                       // agg (fp32)
  float*          Y     = (float*)(ws + szX);                 // relu(conv) fp32
  float*          H1    = (float*)(ws + 2 * szX);             // layer-1 out
  unsigned short* Xbf   = (unsigned short*)(ws + 3 * szX);          // bf16 agg/pooled
  unsigned short* Ybf   = (unsigned short*)(ws + 3 * szX + szXb);   // bf16 relu
  unsigned short* W1f   = (unsigned short*)(ws + 3 * szX + 2 * szXb);
  unsigned short* rW1f  = W1f  + DHID * DHID;
  unsigned short* W2f   = rW1f + DHID * DHID;
  unsigned short* rW2f  = W2f  + DHID * DHID;
  unsigned short* linWf = rW2f + DHID * DHID;

  // ---- pack weights into bf16 WMMA B-fragments ----
  {
    int tot = DHID * DHID;
    int blk = (tot + 255) / 256;
    pack_b_kernel<<<blk, 256, 0, stream>>>(W1,  W1f,  DHID);
    pack_b_kernel<<<blk, 256, 0, stream>>>(rW1, rW1f, DHID);
    pack_b_kernel<<<blk, 256, 0, stream>>>(W2,  W2f,  DHID);
    pack_b_kernel<<<blk, 256, 0, stream>>>(rW2, rW2f, DHID);
    int totL = DHID * NL;
    pack_b_kernel<<<(totL + 255) / 256, 256, 0, stream>>>(linW, linWf, NL);
  }

  const int nElem  = M * DHID;
  const int n4     = nElem / 4;
  const int cpBlk  = (n4 + 255) / 256;
  const int scBlk  = (E * 32 + 255) / 256;
  const int mtiles = (M + 15) / 16;

  // gemm grids: 8 waves/block, one (mtile, ntile-group-of-8) task per wave
  const int tasksD = mtiles * ((DHID / 16) / 8);   // D=128 -> 1 group
  const int tasksL = mtiles * ((NL   / 16) / 8);   // N=1024 -> 8 groups
  const int gD = (tasksD + 7) / 8;
  const int gL = (tasksL + 7) / 8;

  // ---- layer 1 ----
  scale_copy_kernel<<<cpBlk, 256, 0, stream>>>((const float4*)inputs, eps1,
                                               (float4*)X, n4);
  scatter_add_kernel<<<scBlk, 256, 0, stream>>>(inputs, src, dst, X, E);
  to_bf16_kernel<<<cpBlk, 256, 0, stream>>>((const float4*)X, (uint2*)Xbf, n4);
  gemm_wmma_kernel<0, 8><<<gD, 256, 0, stream>>>(Xbf, W1f, b1, nullptr, nullptr,
                                                 Y, Ybf, M, DHID);
  gemm_wmma_kernel<1, 8><<<gD, 256, 0, stream>>>(Ybf, rW1f, rb1, Y, nullptr,
                                                 H1, nullptr, M, DHID);

  // ---- layer 2 (pool fused into residual epilogue; pooled kept bf16-only) ----
  scale_copy_kernel<<<cpBlk, 256, 0, stream>>>((const float4*)H1, eps2,
                                               (float4*)X, n4);
  scatter_add_kernel<<<scBlk, 256, 0, stream>>>(H1, src, dst, X, E);
  to_bf16_kernel<<<cpBlk, 256, 0, stream>>>((const float4*)X, (uint2*)Xbf, n4);
  gemm_wmma_kernel<0, 8><<<gD, 256, 0, stream>>>(Xbf, W2f, b2, nullptr, nullptr,
                                                 Y, Ybf, M, DHID);
  gemm_wmma_kernel<2, 8><<<gD, 256, 0, stream>>>(Ybf, rW2f, rb2, Y, H1,
                                                 nullptr, Xbf, M, DHID);

  // ---- label head: out = pooled @ linW + linb ----
  gemm_wmma_kernel<3, 8><<<gL, 256, 0, stream>>>(Xbf, linWf, linb, nullptr,
                                                 nullptr, (float*)d_out,
                                                 nullptr, M, NL);
}
